// InstanceSegmentationHead_1666447310905
// MI455X (gfx1250) — compile-verified, matrix operand order checked by reading, not run on hardware
//
#include <hip/hip_runtime.h>
#include <math.h>

// ---------------- constants ----------------
#define BATCH 2
#define NPOS 256            // 16x16 tokens
#define CIN 384
#define HH 224
#define WW 224
#define HWSZ (224*224)      // 50176
#define PADW 226
#define PADSZ (226*226)     // 51076
#define NPROTO 64
#define EMB 32

typedef __attribute__((ext_vector_type(16))) _Float16 v16h;
typedef __attribute__((ext_vector_type(8)))  float    v8f;

// =====================================================================
// Generic strided WMMA GEMM:  C(m,n) = act( alpha*sum_k A(m,k)B(k,n) [+bias] [+C_old] )
//   m decomposed as y=m/Wm, x=m%Wm
//   A(m,k) = A[y*sa_y + x*sa_x + k*sa_k]
//   B(k,n) = B[k*sb_k + n*sb_n]
//   C(m,n) = C[y*sc_y + x*sc_x + n*sc_n]
// Requires: M%16==0, N%(16*NF)==0, K%32==0. One wave32 per 16x(16*NF) tile.
// LDS tiles are stored in *fragment order*: lane L's v16h fragment is the
// 32 contiguous bytes at [L*16], so gather is two ds_load_b128 per fragment.
//   WMMA 16-bit operand layout (wave32): element e of lane (16*g + r) holds
//   K = (e&7) + 8*g + 16*(e>>3); row r = M (for A) or N (for B).
// Staging is two-phase (batch all global loads, then convert+store to LDS)
// so the 16 loads per tile stay in flight together instead of serializing
// on per-element s_wait_loadcnt.
// =====================================================================
template <int NF>
__global__ __launch_bounds__(32)
void k_wmma_gemm(const float* __restrict__ A, const float* __restrict__ Bw,
                 float* __restrict__ C, const float* __restrict__ bias,
                 int M, int N, int K, int Wm,
                 long sa_y, long sa_x, long sa_k,
                 long sb_k, long sb_n,
                 long sc_y, long sc_x, long sc_n,
                 float alpha, int beta, int relu)
{
    __shared__ __align__(32) _Float16 As[512];
    __shared__ __align__(32) _Float16 Bs[512 * NF];
    const int lane = threadIdx.x;          // 0..31
    const int m0 = blockIdx.x * 16;
    const int n0 = blockIdx.y * 16 * NF;

    // ---- hoisted per-lane staging addresses (k-invariant) ----
    long aoff[16]; int adst[16];
    long boff[16]; int bdst[16];
#pragma unroll
    for (int t = 0; t < 16; ++t) {
        int f = lane * 16 + t;                  // 0..511
        int ar = f >> 5, ac = f & 31;           // A tile: row(m), col(k)
        int m = m0 + ar;
        aoff[t] = (long)(m / Wm) * sa_y + (long)(m % Wm) * sa_x + (long)ac * sa_k;
        adst[t] = ((((ac >> 3) & 1) << 4) + ar) * 16 + ((ac & 7) + ((ac >> 4) << 3));
        int bk = f >> 4, bn = f & 15;           // B tile: row(k), col(n)
        boff[t] = (long)bk * sb_k + (long)(n0 + bn) * sb_n;
        bdst[t] = ((((bk >> 3) & 1) << 4) + bn) * 16 + ((bk & 7) + ((bk >> 4) << 3));
    }

    v8f acc[NF];
#pragma unroll
    for (int nf = 0; nf < NF; ++nf) acc[nf] = (v8f){0.f,0.f,0.f,0.f,0.f,0.f,0.f,0.f};

    for (int k0 = 0; k0 < K; k0 += 32) {
        const long ka = (long)k0 * sa_k;
        const long kb = (long)k0 * sb_k;
        if (k0 + 32 < K)  // gfx1250 global_prefetch_b8 hint for the next A chunk
            __builtin_prefetch(&A[aoff[0] + ka + 32 * sa_k], 0, 1);
        // ---- phase 1: batch global loads (MLP), phase 2: convert + LDS store ----
        float av[16];
#pragma unroll
        for (int t = 0; t < 16; ++t) av[t] = A[aoff[t] + ka];
#pragma unroll
        for (int t = 0; t < 16; ++t) As[adst[t]] = (_Float16)av[t];
#pragma unroll
        for (int nf = 0; nf < NF; ++nf) {
            float bv[16];
#pragma unroll
            for (int t = 0; t < 16; ++t)
                bv[t] = Bw[boff[t] + kb + (long)(nf * 16) * sb_n];
#pragma unroll
            for (int t = 0; t < 16; ++t) Bs[nf * 512 + bdst[t]] = (_Float16)bv[t];
        }
        __syncthreads();
        v16h af = *reinterpret_cast<const v16h*>(As + lane * 16);
#pragma unroll
        for (int nf = 0; nf < NF; ++nf) {
            v16h bf = *reinterpret_cast<const v16h*>(Bs + nf * 512 + lane * 16);
            acc[nf] = __builtin_amdgcn_wmma_f32_16x16x32_f16(
                          false, af, false, bf, (short)0, acc[nf], false, false);
        }
        __syncthreads();
    }

    // epilogue: D layout -> VGPR v holds M = v + 8*(lane>>4), N = lane&15
    const int nn = lane & 15, g = lane >> 4;
#pragma unroll
    for (int nf = 0; nf < NF; ++nf) {
#pragma unroll
        for (int v = 0; v < 8; ++v) {
            int m = m0 + g * 8 + v;
            int n = n0 + nf * 16 + nn;
            long caddr = (long)(m / Wm) * sc_y + (long)(m % Wm) * sc_x + (long)n * sc_n;
            float val = acc[nf][v] * alpha;
            if (bias) val += bias[n];
            if (beta) val += C[caddr];
            if (relu) val = fmaxf(val, 0.f);
            C[caddr] = val;
        }
    }
}

// =====================================================================
// elementwise / support kernels
// =====================================================================
__global__ void k_fill(float* p, long n, float v) {
    long i = (long)blockIdx.x * blockDim.x + threadIdx.x;
    if (i < n) p[i] = v;
}

// zero-pad (C,H,W) -> (C,H+2,W+2)
__global__ void k_pad(const float* __restrict__ src, float* __restrict__ dst,
                      int C, int H, int W) {
    int Wp = W + 2, Hp = H + 2;
    long tot = (long)C * Hp * Wp;
    long i = (long)blockIdx.x * blockDim.x + threadIdx.x;
    if (i >= tot) return;
    int x = (int)(i % Wp); int y = (int)((i / Wp) % Hp); int c = (int)(i / ((long)Wp * Hp));
    float v = 0.f;
    if (y >= 1 && y <= H && x >= 1 && x <= W)
        v = src[(long)c * H * W + (long)(y - 1) * W + (x - 1)];
    dst[i] = v;
}

// bilinear align_corners resize (C,Hs,Ws) -> (C,Hd,Wd), strided dst
__global__ void k_resize(const float* __restrict__ src, int C, int Hs, int Ws,
                         float* __restrict__ dst, int Hd, int Wd,
                         long dcs, long drs) {
    long tot = (long)C * Hd * Wd;
    long i = (long)blockIdx.x * blockDim.x + threadIdx.x;
    if (i >= tot) return;
    int x = (int)(i % Wd); int y = (int)((i / Wd) % Hd); int c = (int)(i / ((long)Wd * Hd));
    float sh = (Hd > 1) ? (float)(Hs - 1) / (float)(Hd - 1) : 0.f;
    float sw = (Wd > 1) ? (float)(Ws - 1) / (float)(Wd - 1) : 0.f;
    float fy = y * sh, fx = x * sw;
    int y0 = (int)floorf(fy); if (y0 > Hs - 2) y0 = Hs - 2; if (y0 < 0) y0 = 0;
    int x0 = (int)floorf(fx); if (x0 > Ws - 2) x0 = Ws - 2; if (x0 < 0) x0 = 0;
    float ty = fy - (float)y0, tx = fx - (float)x0;
    const float* sp = src + (long)c * Hs * Ws;
    float v00 = sp[y0 * Ws + x0],       v01 = sp[y0 * Ws + x0 + 1];
    float v10 = sp[(y0 + 1) * Ws + x0], v11 = sp[(y0 + 1) * Ws + x0 + 1];
    float v = (1.f - ty) * ((1.f - tx) * v00 + tx * v01)
            +        ty  * ((1.f - tx) * v10 + tx * v11);
    dst[(long)c * dcs + (long)y * drs + x] = v;
}

// in-place BN+ReLU on the interior of a padded buffer
__global__ void k_bnrelu(float* buf, int C, long cs, long rs,
                         const float* g, const float* bt, const float* mn, const float* vr) {
    long tot = (long)C * HH * WW;
    long i = (long)blockIdx.x * blockDim.x + threadIdx.x;
    if (i >= tot) return;
    int x = (int)(i % WW); int y = (int)((i / WW) % HH); int c = (int)(i / HWSZ);
    long a = (long)c * cs + (long)y * rs + x;
    float sc = g[c] * rsqrtf(vr[c] + 1e-5f);
    float v = (buf[a] - mn[c]) * sc + bt[c];
    buf[a] = fmaxf(v, 0.f);
}

// edge map from pe gradients + global mean accumulation
__global__ void k_edge(const float* __restrict__ pe, float* __restrict__ edge, float* sum) {
    int idx = blockIdx.x * blockDim.x + threadIdx.x;
    int b = blockIdx.y;
    if (idx >= HWSZ) return;
    int x = idx % WW, y = idx / WW;
    const float* pb = pe + (long)b * EMB * HWSZ;
    int x0 = (x < WW - 1) ? x : WW - 2;
    int y0 = (y < HH - 1) ? y : HH - 2;
    float gx = 0.f, gy = 0.f;
    for (int e = 0; e < EMB; ++e) {
        const float* ch = pb + (long)e * HWSZ;
        gx += fabsf(ch[y * WW + x0] - ch[y * WW + x0 + 1]);
        gy += fabsf(ch[y0 * WW + x] - ch[(y0 + 1) * WW + x]);
    }
    float ev = 0.5f * (gx + gy) * (1.0f / EMB);
    edge[(long)b * HWSZ + idx] = ev;
    atomicAdd(sum, ev * (1.0f / ((float)BATCH * HWSZ)));
}

__global__ void k_mask(const float* __restrict__ edge, const float* mean,
                       float* __restrict__ mask, long n) {
    long i = (long)blockIdx.x * blockDim.x + threadIdx.x;
    if (i < n) mask[i] = (edge[i] > *mean) ? 1.f : 0.f;
}

// 3x3 mean with implicit zero pad, always /9
__global__ void k_avg3(const float* __restrict__ src, float* __restrict__ dst) {
    long tot = (long)BATCH * NPROTO * HWSZ;
    long i = (long)blockIdx.x * blockDim.x + threadIdx.x;
    if (i >= tot) return;
    int x = (int)(i % WW); int y = (int)((i / WW) % HH);
    long plane = i / HWSZ;
    const float* sp = src + plane * HWSZ;
    float s = 0.f;
    for (int dy = -1; dy <= 1; ++dy)
        for (int dx = -1; dx <= 1; ++dx) {
            int yy = y + dy, xx = x + dx;
            if (yy >= 0 && yy < HH && xx >= 0 && xx < WW) s += sp[yy * WW + xx];
        }
    dst[i] = s * (1.0f / 9.0f);
}

__global__ void k_combine(float* __restrict__ sim, const float* __restrict__ tmp,
                          const float* __restrict__ mask) {
    long tot = (long)BATCH * NPROTO * HWSZ;
    long i = (long)blockIdx.x * blockDim.x + threadIdx.x;
    if (i >= tot) return;
    long pix = i % HWSZ;
    long b = i / ((long)NPROTO * HWSZ);
    float m = mask[b * HWSZ + pix];
    sim[i] = m * sim[i] + (1.f - m) * tmp[i];
}

// softmax-entropy over prototypes, mean-accumulated via atomics
__global__ void k_entropy(const float* __restrict__ sim, float* ent_out) {
    int idx = blockIdx.x * blockDim.x + threadIdx.x;
    int b = blockIdx.y;
    if (idx >= HWSZ) return;
    const float* s = sim + (long)b * NPROTO * HWSZ + idx;
    float mx = -1e30f;
    for (int p = 0; p < NPROTO; ++p) mx = fmaxf(mx, s[(long)p * HWSZ]);
    float den = 0.f;
    for (int p = 0; p < NPROTO; ++p) den += expf(s[(long)p * HWSZ] - mx);
    float inv = 1.f / den, ent = 0.f;
    for (int p = 0; p < NPROTO; ++p) {
        float q = expf(s[(long)p * HWSZ] - mx) * inv;
        ent -= q * logf(q + 1e-8f);
    }
    atomicAdd(ent_out, ent * (1.0f / ((float)BATCH * HWSZ)));
}

__global__ void k_softmax(const float* __restrict__ sim, float* __restrict__ act) {
    int idx = blockIdx.x * blockDim.x + threadIdx.x;
    int b = blockIdx.y;
    if (idx >= HWSZ) return;
    const float* s = sim + (long)b * NPROTO * HWSZ + idx;
    float* a = act + (long)b * NPROTO * HWSZ + idx;
    float mx = -1e30f;
    for (int p = 0; p < NPROTO; ++p) mx = fmaxf(mx, s[(long)p * HWSZ]);
    float den = 0.f;
    for (int p = 0; p < NPROTO; ++p) den += expf(s[(long)p * HWSZ] - mx);
    float inv = 1.f / den;
    for (int p = 0; p < NPROTO; ++p) a[(long)p * HWSZ] = expf(s[(long)p * HWSZ] - mx) * inv;
}

__global__ void k_bb(const float* pb, const float* bscale, float* bbuf, float* outbb) {
    int p = threadIdx.x;
    if (p < NPROTO) {
        float v = 1.f + tanhf(pb[p]) * bscale[0];
        bbuf[p] = v;
        outbb[p] = v;            // batch 0
        outbb[NPROTO + p] = v;   // batch 1 (broadcast)
    }
}

__global__ void k_basegain(const float* __restrict__ act, const float* __restrict__ bb,
                           float* __restrict__ gain0) {
    int idx = blockIdx.x * blockDim.x + threadIdx.x;
    int b = blockIdx.y;
    if (idx >= HWSZ) return;
    const float* a = act + (long)b * NPROTO * HWSZ + idx;
    float s = 0.f;
    for (int p = 0; p < NPROTO; ++p) s += a[(long)p * HWSZ] * bb[p];
    gain0[(long)b * HWSZ + idx] = s;
}

// grouped head convs (groups=64), tiny K -> direct VALU, bounds-checked
__global__ void k_head1(const float* __restrict__ act, const float* __restrict__ w,
                        const float* __restrict__ bias, const float* g, const float* bt,
                        const float* mn, const float* vr, float* __restrict__ out) {
    long tot = (long)512 * HWSZ;
    long i = (long)blockIdx.x * blockDim.x + threadIdx.x;
    if (i >= tot) return;
    int x = (int)(i % WW); int y = (int)((i / WW) % HH); int oc = (int)(i / HWSZ);
    int p = oc >> 3;
    const float* in = act + (long)p * HWSZ;
    float v = bias[oc];
    for (int dy = 0; dy < 3; ++dy)
        for (int dx = 0; dx < 3; ++dx) {
            int yy = y + dy - 1, xx = x + dx - 1;
            if (yy >= 0 && yy < HH && xx >= 0 && xx < WW)
                v += in[yy * WW + xx] * w[oc * 9 + dy * 3 + dx];
        }
    float sc = g[oc] * rsqrtf(vr[oc] + 1e-5f);
    v = (v - mn[oc]) * sc + bt[oc];
    out[i] = fmaxf(v, 0.f);
}

__global__ void k_head2(const float* __restrict__ h1, const float* __restrict__ w,
                        const float* __restrict__ bias, const float* g, const float* bt,
                        const float* mn, const float* vr, float* __restrict__ out) {
    long tot = (long)512 * HWSZ;
    long i = (long)blockIdx.x * blockDim.x + threadIdx.x;
    if (i >= tot) return;
    int x = (int)(i % WW); int y = (int)((i / WW) % HH); int oc = (int)(i / HWSZ);
    int p = oc >> 3;
    float v = bias[oc];
    for (int ic = 0; ic < 8; ++ic) {
        const float* in = h1 + (long)(p * 8 + ic) * HWSZ;
        const float* wk = w + oc * 72 + ic * 9;
        for (int dy = 0; dy < 3; ++dy)
            for (int dx = 0; dx < 3; ++dx) {
                int yy = y + dy - 1, xx = x + dx - 1;
                if (yy >= 0 && yy < HH && xx >= 0 && xx < WW)
                    v += in[yy * WW + xx] * wk[dy * 3 + dx];
            }
    }
    float sc = g[oc] * rsqrtf(vr[oc] + 1e-5f);
    v = (v - mn[oc]) * sc + bt[oc];
    out[i] = fmaxf(v, 0.f);
}

__global__ void k_head3(const float* __restrict__ h2, const float* __restrict__ w,
                        const float* __restrict__ bias, float* __restrict__ res) {
    long tot = (long)NPROTO * HWSZ;
    long i = (long)blockIdx.x * blockDim.x + threadIdx.x;
    if (i >= tot) return;
    int x = (int)(i % WW); int y = (int)((i / WW) % HH); int p = (int)(i / HWSZ);
    float v = bias[p];
    for (int ic = 0; ic < 8; ++ic) {
        const float* in = h2 + (long)(p * 8 + ic) * HWSZ;
        const float* wk = w + p * 72 + ic * 9;
        for (int dy = 0; dy < 3; ++dy)
            for (int dx = 0; dx < 3; ++dx) {
                int yy = y + dy - 1, xx = x + dx - 1;
                if (yy >= 0 && yy < HH && xx >= 0 && xx < WW)
                    v += in[yy * WW + xx] * wk[dy * 3 + dx];
            }
    }
    res[i] = v;
}

__global__ void k_final(const float* __restrict__ act, const float* __restrict__ res,
                        const float* __restrict__ gain0, const float* rscale,
                        float* __restrict__ weighted, float* __restrict__ ibm) {
    int idx = blockIdx.x * blockDim.x + threadIdx.x;
    if (idx >= HWSZ) return;
    float ws = 0.f;
    for (int p = 0; p < NPROTO; ++p)
        ws += act[(long)p * HWSZ + idx] * res[(long)p * HWSZ + idx];
    weighted[idx] = ws;
    float gain = gain0[idx] * (1.f + tanhf(ws) * rscale[0]);
    ibm[idx] = fminf(fmaxf(gain, 0.5f), 2.0f);
}

// =====================================================================
// host side
// =====================================================================
static inline void gemm(hipStream_t s, const float* A, const float* Bw, float* C,
                        const float* bias, int M, int N, int K, int Wm,
                        long say, long sax, long sak, long sbk, long sbn,
                        long scy, long scx, long scn,
                        float alpha, int beta, int relu) {
    if (N % 64 == 0) {
        dim3 g(M / 16, N / 64);
        k_wmma_gemm<4><<<g, dim3(32), 0, s>>>(A, Bw, C, bias, M, N, K, Wm,
                                              say, sax, sak, sbk, sbn, scy, scx, scn,
                                              alpha, beta, relu);
    } else if (N % 32 == 0) {
        dim3 g(M / 16, N / 32);
        k_wmma_gemm<2><<<g, dim3(32), 0, s>>>(A, Bw, C, bias, M, N, K, Wm,
                                              say, sax, sak, sbk, sbn, scy, scx, scn,
                                              alpha, beta, relu);
    } else {
        dim3 g(M / 16, N / 16);
        k_wmma_gemm<1><<<g, dim3(32), 0, s>>>(A, Bw, C, bias, M, N, K, Wm,
                                              say, sax, sak, sbk, sbn, scy, scx, scn,
                                              alpha, beta, relu);
    }
}
static inline dim3 g1(long n) { return dim3((unsigned)((n + 255) / 256)); }

extern "C" void kernel_launch(void* const* d_in, const int* in_sizes, int n_in,
                              void* d_out, int out_size, void* d_ws, size_t ws_size,
                              hipStream_t stream) {
    // ---- inputs ----
    const float* tokens[4] = {(const float*)d_in[0], (const float*)d_in[1],
                              (const float*)d_in[2], (const float*)d_in[3]};
    const float* proj_w[4] = {(const float*)d_in[4], (const float*)d_in[6],
                              (const float*)d_in[8], (const float*)d_in[10]};
    const float* proj_b[4] = {(const float*)d_in[5], (const float*)d_in[7],
                              (const float*)d_in[9], (const float*)d_in[11]};
    const float* up_w0 = (const float*)d_in[12]; const float* up_b0 = (const float*)d_in[13];
    const float* up_w1 = (const float*)d_in[14]; const float* up_b1 = (const float*)d_in[15];
    const float* down_w3 = (const float*)d_in[16]; const float* down_b3 = (const float*)d_in[17];
    const float* fuse_w = (const float*)d_in[18]; const float* fuse_b = (const float*)d_in[19];
    const float* pe_w1 = (const float*)d_in[20]; const float* pe_b1 = (const float*)d_in[21];
    const float* pe_bn_g = (const float*)d_in[22]; const float* pe_bn_b = (const float*)d_in[23];
    const float* pe_bn_m = (const float*)d_in[24]; const float* pe_bn_v = (const float*)d_in[25];
    const float* pe_w2 = (const float*)d_in[26]; const float* pe_b2 = (const float*)d_in[27];
    const float* prototypes = (const float*)d_in[28];
    const float* proto_bias = (const float*)d_in[29];
    const float* bias_scale = (const float*)d_in[30];
    const float* smooth_w = (const float*)d_in[31]; const float* smooth_b = (const float*)d_in[32];
    const float* head_w1 = (const float*)d_in[33]; const float* head_b1 = (const float*)d_in[34];
    const float* hbn1_g = (const float*)d_in[35]; const float* hbn1_b = (const float*)d_in[36];
    const float* hbn1_m = (const float*)d_in[37]; const float* hbn1_v = (const float*)d_in[38];
    const float* head_w2 = (const float*)d_in[39]; const float* head_b2 = (const float*)d_in[40];
    const float* hbn2_g = (const float*)d_in[41]; const float* hbn2_b = (const float*)d_in[42];
    const float* hbn2_m = (const float*)d_in[43]; const float* hbn2_v = (const float*)d_in[44];
    const float* head_w3 = (const float*)d_in[45]; const float* head_b3 = (const float*)d_in[46];
    const float* res_scale = (const float*)d_in[47];
    (void)in_sizes; (void)n_in; (void)out_size; (void)ws_size;

    float* out = (float*)d_out;
    const long OFF_IBM = 0;
    const long OFF_ACT = (long)BATCH * HWSZ;                        // 100352
    const long OFF_BB  = OFF_ACT + (long)BATCH * NPROTO * HWSZ;     // 6522880
    const long OFF_W   = OFF_BB + BATCH * NPROTO;                   // 6523008
    const long OFF_PE  = OFF_W + (long)BATCH * HWSZ;                // 6623360
    const long OFF_ENT = OFF_PE + (long)BATCH * EMB * HWSZ;         // 9834624

    // ---- scratch layout (floats) ----
    float* ws = (float*)d_ws;
    size_t cur = 0;
    auto alloc = [&](size_t n) { float* p = ws + cur; cur += n; return p; };
    const int OC[4] = {256, 512, 1024, 1024};
    float* feats[4];
    for (int i = 0; i < 4; ++i) feats[i] = alloc((size_t)BATCH * OC[i] * NPOS);
    float* f3pad = alloc((size_t)BATCH * 1024 * 18 * 18);
    float* f3    = alloc((size_t)BATCH * 1024 * 64);
    float* f1    = alloc((size_t)BATCH * 512 * 1024);
    const int CF = 2816;
    float* fused  = alloc((size_t)BATCH * CF * 4096);
    float* path1  = alloc((size_t)BATCH * 256 * 4096);
    float* p1pad  = alloc((size_t)BATCH * 256 * PADSZ);
    float* pe1pad = alloc((size_t)BATCH * 128 * PADSZ);
    float* sim0   = alloc((size_t)BATCH * NPROTO * HWSZ);
    float* simt   = alloc((size_t)BATCH * NPROTO * HWSZ);
    float* simpad = alloc((size_t)BATCH * NPROTO * PADSZ);
    float* sims   = alloc((size_t)BATCH * NPROTO * HWSZ);
    float* edgeb  = alloc((size_t)BATCH * HWSZ);
    float* maskb  = alloc((size_t)BATCH * HWSZ);
    float* esum   = alloc(1);
    float* bbuf   = alloc(NPROTO);
    float* gain0  = alloc((size_t)BATCH * HWSZ);
    float* resb   = alloc((size_t)NPROTO * HWSZ);   // per-batch reuse
    float* h1     = alloc((size_t)512 * HWSZ);      // per-batch reuse
    float* h2     = alloc((size_t)512 * HWSZ);      // per-batch reuse

    // ---- init accumulators / padded borders ----
    k_fill<<<g1(1), 256, 0, stream>>>(esum, 1, 0.f);
    k_fill<<<g1(1), 256, 0, stream>>>(out + OFF_ENT, 1, 0.f);
    k_fill<<<g1((long)BATCH * 256 * PADSZ), 256, 0, stream>>>(p1pad, (long)BATCH * 256 * PADSZ, 0.f);
    k_fill<<<g1((long)BATCH * 128 * PADSZ), 256, 0, stream>>>(pe1pad, (long)BATCH * 128 * PADSZ, 0.f);

    for (int b = 0; b < BATCH; ++b) {
        // 1x1 token projections: (256 x 384) x (384 x OC)
        for (int i = 0; i < 4; ++i)
            gemm(stream, tokens[i] + (size_t)b * NPOS * CIN, proj_w[i],
                 feats[i] + (size_t)b * OC[i] * NPOS, proj_b[i],
                 NPOS, OC[i], CIN, NPOS, 0, CIN, 1, 1, CIN, 0, 1, NPOS, 1.f, 0, 0);

        // convT 4x (f0) -> fused slab [0:256)
        for (int i = 0; i < 4; ++i)
            for (int j = 0; j < 4; ++j)
                gemm(stream, feats[0] + (size_t)b * 256 * NPOS, up_w0 + i * 4 + j,
                     fused + (size_t)b * CF * 4096 + i * 64 + j, up_b0,
                     NPOS, 256, 256, 16, 16, 1, NPOS, 4096, 16, 256, 4, 4096, 1.f, 0, 0);
        // convT 2x (f1) -> f1 buffer (32x32)
        for (int i = 0; i < 2; ++i)
            for (int j = 0; j < 2; ++j)
                gemm(stream, feats[1] + (size_t)b * 512 * NPOS, up_w1 + i * 2 + j,
                     f1 + (size_t)b * 512 * 1024 + i * 32 + j, up_b1,
                     NPOS, 512, 512, 16, 16, 1, NPOS, 2048, 4, 64, 2, 1024, 1.f, 0, 0);

        // stride-2 3x3 down conv (f3): pad feats3, 9 accumulated taps
        k_pad<<<g1((long)1024 * 18 * 18), 256, 0, stream>>>(
            feats[3] + (size_t)b * 1024 * NPOS, f3pad + (size_t)b * 1024 * 324, 1024, 16, 16);
        int tap = 0;
        for (int dy = 0; dy < 3; ++dy)
            for (int dx = 0; dx < 3; ++dx, ++tap)
                gemm(stream, f3pad + (size_t)b * 1024 * 324 + dy * 18 + dx,
                     down_w3 + dy * 3 + dx, f3 + (size_t)b * 1024 * 64,
                     tap == 0 ? down_b3 : nullptr,
                     64, 1024, 1024, 8, 36, 2, 324, 9, 9216, 8, 1, 64, 1.f, tap ? 1 : 0, 0);

        // bilinear resizes into fused slabs
        k_resize<<<g1((long)512 * 4096), 256, 0, stream>>>(
            f1 + (size_t)b * 512 * 1024, 512, 32, 32,
            fused + (size_t)b * CF * 4096 + (size_t)256 * 4096, 64, 64, 4096, 64);
        k_resize<<<g1((long)1024 * 4096), 256, 0, stream>>>(
            feats[2] + (size_t)b * 1024 * NPOS, 1024, 16, 16,
            fused + (size_t)b * CF * 4096 + (size_t)768 * 4096, 64, 64, 4096, 64);
        k_resize<<<g1((long)1024 * 4096), 256, 0, stream>>>(
            f3 + (size_t)b * 1024 * 64, 1024, 8, 8,
            fused + (size_t)b * CF * 4096 + (size_t)1792 * 4096, 64, 64, 4096, 64);

        // fuse 1x1 conv + ReLU: M=4096 K=2816 N=256
        gemm(stream, fused + (size_t)b * CF * 4096, fuse_w, path1 + (size_t)b * 256 * 4096,
             fuse_b, 4096, 256, CF, 64, 64, 1, 4096, 1, CF, 64, 1, 4096, 1.f, 0, 1);

        // upsample path1 64->224 directly into padded interior
        k_resize<<<g1((long)256 * HWSZ), 256, 0, stream>>>(
            path1 + (size_t)b * 256 * 4096, 256, 64, 64,
            p1pad + (size_t)b * 256 * PADSZ + 227, HH, WW, PADSZ, PADW);

        // pe conv1 (3x3, 256->128): 9 WMMA taps into padded interior
        tap = 0;
        for (int dy = 0; dy < 3; ++dy)
            for (int dx = 0; dx < 3; ++dx, ++tap)
                gemm(stream, p1pad + (size_t)b * 256 * PADSZ + dy * PADW + dx,
                     pe_w1 + dy * 3 + dx, pe1pad + (size_t)b * 128 * PADSZ + 227,
                     tap == 0 ? pe_b1 : nullptr,
                     HWSZ, 128, 256, WW, PADW, 1, PADSZ, 9, 2304, PADW, 1, PADSZ,
                     1.f, tap ? 1 : 0, 0);
        k_bnrelu<<<g1((long)128 * HWSZ), 256, 0, stream>>>(
            pe1pad + (size_t)b * 128 * PADSZ + 227, 128, PADSZ, PADW,
            pe_bn_g, pe_bn_b, pe_bn_m, pe_bn_v);

        // pe conv2 (3x3, 128->32) straight into d_out pe slice
        float* peb = out + OFF_PE + (size_t)b * EMB * HWSZ;
        tap = 0;
        for (int dy = 0; dy < 3; ++dy)
            for (int dx = 0; dx < 3; ++dx, ++tap)
                gemm(stream, pe1pad + (size_t)b * 128 * PADSZ + dy * PADW + dx,
                     pe_w2 + dy * 3 + dx, peb, tap == 0 ? pe_b2 : nullptr,
                     HWSZ, EMB, 128, WW, PADW, 1, PADSZ, 9, 1152, WW, 1, HWSZ,
                     1.f, tap ? 1 : 0, 0);

        // prototype similarity: M=50176 K=32 N=64, alpha = 1/sqrt(32)
        gemm(stream, peb, prototypes, sim0 + (size_t)b * NPROTO * HWSZ, nullptr,
             HWSZ, NPROTO, EMB, WW, WW, 1, HWSZ, 1, EMB, WW, 1, HWSZ,
             0.17677669529663689f, 0, 0);
    }

    // edge map + mean + mask
    k_edge<<<dim3((HWSZ + 255) / 256, BATCH), 256, 0, stream>>>(out + OFF_PE, edgeb, esum);
    k_mask<<<g1((long)BATCH * HWSZ), 256, 0, stream>>>(edgeb, esum, maskb, (long)BATCH * HWSZ);

    // two edge-aware smoothing steps
    for (int it = 0; it < 2; ++it) {
        k_avg3<<<g1((long)BATCH * NPROTO * HWSZ), 256, 0, stream>>>(sim0, simt);
        k_combine<<<g1((long)BATCH * NPROTO * HWSZ), 256, 0, stream>>>(sim0, simt, maskb);
    }

    // entropy of softmax(sim)
    k_entropy<<<dim3((HWSZ + 255) / 256, BATCH), 256, 0, stream>>>(sim0, out + OFF_ENT);

    // smooth conv (3x3, 64->64) via padded 9-tap WMMA
    for (int b = 0; b < BATCH; ++b) {
        k_pad<<<g1((long)NPROTO * PADSZ), 256, 0, stream>>>(
            sim0 + (size_t)b * NPROTO * HWSZ, simpad + (size_t)b * NPROTO * PADSZ,
            NPROTO, HH, WW);
        int tap = 0;
        for (int dy = 0; dy < 3; ++dy)
            for (int dx = 0; dx < 3; ++dx, ++tap)
                gemm(stream, simpad + (size_t)b * NPROTO * PADSZ + dy * PADW + dx,
                     smooth_w + dy * 3 + dx, sims + (size_t)b * NPROTO * HWSZ,
                     tap == 0 ? smooth_b : nullptr,
                     HWSZ, NPROTO, NPROTO, WW, PADW, 1, PADSZ, 9, 576, WW, 1, HWSZ,
                     1.f, tap ? 1 : 0, 0);
    }

    // act = softmax(sims) -> d_out act slice
    k_softmax<<<dim3((HWSZ + 255) / 256, BATCH), 256, 0, stream>>>(sims, out + OFF_ACT);

    // base biases + base gain
    k_bb<<<dim3(1), dim3(64), 0, stream>>>(proto_bias, bias_scale, bbuf, out + OFF_BB);
    k_basegain<<<dim3((HWSZ + 255) / 256, BATCH), 256, 0, stream>>>(out + OFF_ACT, bbuf, gain0);

    // grouped residual heads + final gain, per batch (reuse h1/h2/res scratch)
    for (int b = 0; b < BATCH; ++b) {
        const float* actb = out + OFF_ACT + (size_t)b * NPROTO * HWSZ;
        k_head1<<<g1((long)512 * HWSZ), 256, 0, stream>>>(
            actb, head_w1, head_b1, hbn1_g, hbn1_b, hbn1_m, hbn1_v, h1);
        k_head2<<<g1((long)512 * HWSZ), 256, 0, stream>>>(
            h1, head_w2, head_b2, hbn2_g, hbn2_b, hbn2_m, hbn2_v, h2);
        k_head3<<<g1((long)NPROTO * HWSZ), 256, 0, stream>>>(h2, head_w3, head_b3, resb);
        k_final<<<g1(HWSZ), 256, 0, stream>>>(actb, resb, gain0 + (size_t)b * HWSZ,
                                              res_scale, out + OFF_W + (size_t)b * HWSZ,
                                              out + OFF_IBM + (size_t)b * HWSZ);
    }
}
